// ProposedLiquidMambaModel_33440615367070
// MI455X (gfx1250) — matrix operand order repaired and checked by reading, not compiled, not run on hardware
//
#include <hip/hip_runtime.h>
#include <hip/hip_bf16.h>

// ---------------------------------------------------------------------------
// Problem constants (from the reference)
// ---------------------------------------------------------------------------
#define BATCH   8
#define SEQLEN  4096
#define FDIM    64
#define DMODEL  128
#define NSCALE  3
#define NSTATE  16
#define DINNER  256
#define TOTROWS 57344   // 8*(4096+2048+1024)

typedef __attribute__((ext_vector_type(16))) __bf16 v16bf;
typedef __attribute__((ext_vector_type(8)))  float  v8f;
typedef unsigned int u32x4 __attribute__((ext_vector_type(4)));
typedef int          i32x4 __attribute__((ext_vector_type(4)));
typedef int          i32x8 __attribute__((ext_vector_type(8)));

#define DEV __device__ __forceinline__

union ABPack { uint4 u[2]; v16bf v; };

// Row-major A tile (16xK slice, kk = k offset, must be 16B aligned per lane).
// Layout per CDNA5 ISA 7.12.2 (16-bit A 16x32): lane<16 holds K=kk..kk+7 and
// kk+16..kk+23 of row M=lane; lane>=16 holds K=kk+8..15 and kk+24..31.
DEV v16bf load_a(const __bf16* base, long row, int lda, int kk, int lane) {
  const __bf16* p = base + row * (long)lda + kk + ((lane >> 4) * 8);
  ABPack x;
  x.u[0] = *(const uint4*)p;
  x.u[1] = *(const uint4*)(p + 16);
  return x.v;
}

// Pre-swizzled B tile: each 32x16 (KxN) tile is 512 contiguous bf16; lane l
// reads 16 contiguous values = its column slice (packed by pack_w_kernel).
DEV v16bf load_b(const __bf16* pk, int tile, int lane) {
  const uint4* p = (const uint4*)(pk + (long)tile * 512 + lane * 16);
  ABPack x;
  x.u[0] = p[0];
  x.u[1] = p[1];
  return x.v;
}

// Same tile layout but sourced from LDS (emits ds_load_b128).
DEV v16bf load_b_lds(const __bf16* p, int lane) {
  const uint4* q = (const uint4*)(p + lane * 16);
  ABPack x;
  x.u[0] = q[0];
  x.u[1] = q[1];
  return x.v;
}

DEV v8f wmma_bf16(v16bf a, v16bf b, v8f c) {
  return __builtin_amdgcn_wmma_f32_16x16x32_bf16(
      /*neg_a=*/false, a, /*neg_b=*/false, b,
      /*c_mod=*/(short)0, c, /*reuse_a=*/false, /*reuse_b=*/false);
}

DEV int scale_rowoff(int s) { return (s == 0) ? 0 : (s == 1) ? 32768 : 49152; }

// ---------------------------------------------------------------------------
// Tensor Data Mover: 1-D contiguous load of nelem bf16 (nelem <= 65535) from
// global into LDS. D# built per CDNA5 ISA 08_async_tensor §8:
//  group0: [1:0]=count=1, [63:32]=lds_addr, [120:64]=global_addr, [127:126]=2
//  group1: [17:16]=data_size(2B), [79:48]=tensor_dim0, [111:80]=tensor_dim1=1,
//          [127:112]=tile_dim0, [143:128]=tile_dim1=1, [207:160]=dim0_stride
// ---------------------------------------------------------------------------
DEV void tdm_load_b16(unsigned lds_addr, const void* gsrc, unsigned nelem) {
  unsigned long long ga = (unsigned long long)(size_t)gsrc;
  u32x4 g0;
  g0[0] = 1u;                                            // count=1
  g0[1] = lds_addr;                                      // LDS byte address
  g0[2] = (unsigned)ga;                                  // global addr lo
  g0[3] = (unsigned)((ga >> 32) & 0x01FFFFFFu) | (2u << 30);  // hi | type=2
  i32x8 g1;
  g1[0] = 1 << 16;                                       // data_size=2B
  g1[1] = (int)((nelem & 0xFFFFu) << 16);                // tensor_dim0 lo16
  g1[2] = (int)(((nelem >> 16) & 0xFFFFu) | (1u << 16)); // dim0 hi | dim1=1
  g1[3] = (int)((nelem & 0xFFFFu) << 16);                // tile_dim0
  g1[4] = 1;                                             // tile_dim1=1
  g1[5] = (int)nelem;                                    // dim0_stride lo
  g1[6] = 0;
  g1[7] = 0;
  i32x4 z4 = {0, 0, 0, 0};
#if defined(__clang_major__) && (__clang_major__ >= 23)
  i32x8 z8 = {0, 0, 0, 0, 0, 0, 0, 0};
  __builtin_amdgcn_tensor_load_to_lds(g0, g1, z4, z4, z8, 0);
#else
  __builtin_amdgcn_tensor_load_to_lds(g0, g1, z4, z4, 0);
#endif
}

// ---------------------------------------------------------------------------
// Conversion / packing kernels (run once per call; tiny)
// ---------------------------------------------------------------------------
__global__ void cvt_bf16_kernel(const float* __restrict__ src,
                                __bf16* __restrict__ dst, int n) {
  for (int i = blockIdx.x * blockDim.x + threadIdx.x; i < n;
       i += gridDim.x * blockDim.x)
    dst[i] = (__bf16)src[i];
}

// Pack a KxN f32 matrix (row stride rowStride) into bf16 WMMA-B tiles.
// Tile (nt,kt) -> 512 bf16 at ((nt*Ktiles)+kt)*512; within a tile, element
// lane*16+j corresponds to B[kt*32 + (lane>>4)*8 + (j&7) + (j>>3)*16, nt*16 + (lane&15)].
__global__ void pack_w_kernel(const float* __restrict__ src,
                              __bf16* __restrict__ dst, int K, int N,
                              int rowStride) {
  int total = K * N;
  int Ktiles = K >> 5;
  for (int i = blockIdx.x * blockDim.x + threadIdx.x; i < total;
       i += gridDim.x * blockDim.x) {
    int tile = i >> 9;
    int it = i & 511;
    int lane = it >> 4;
    int j = it & 15;
    int nt = tile / Ktiles;
    int kt = tile - nt * Ktiles;
    int n = nt * 16 + (lane & 15);
    int k = kt * 32 + ((lane >> 4) * 8) + (j & 7) + ((j >> 3) * 16);
    dst[i] = (__bf16)src[(long)k * rowStride + n];
  }
}

// ---------------------------------------------------------------------------
// Phase 1: h = x @ Wp + bp   (M=32768, K=64, N=128), bf16 out
// ---------------------------------------------------------------------------
__global__ __launch_bounds__(128) void proj_kernel(
    const __bf16* __restrict__ xbf, const __bf16* __restrict__ wpk,
    const float* __restrict__ bp, __bf16* __restrict__ hbf) {
  int wave = threadIdx.x >> 5, lane = threadIdx.x & 31;
  int mt = blockIdx.x;
  int nt = blockIdx.y * 4 + wave;             // 8 n-tiles total
  long row = (long)mt * 16 + (lane & 15);
  v8f acc = {};
  for (int kt = 0; kt < 2; ++kt) {
    v16bf a = load_a(xbf, row, FDIM, kt * 32, lane);
    v16bf b = load_b(wpk, nt * 2 + kt, lane);
    acc = wmma_bf16(a, b, acc);
  }
  int col = lane & 15, n0 = nt * 16, rb = (lane >> 4) * 8;
  float bv = bp[n0 + col];
  for (int r = 0; r < 8; ++r)
    hbf[((long)mt * 16 + rb + r) * DMODEL + n0 + col] = (__bf16)(acc[r] + bv);
}

// ---------------------------------------------------------------------------
// Phase 2: u = xs @ Win_u + bin_u per scale (strided row gather from h)
// K=128, N=256; bf16 out into the per-scale row block
// ---------------------------------------------------------------------------
__global__ __launch_bounds__(128) void uproj_kernel(
    const __bf16* __restrict__ hbf, const __bf16* __restrict__ winupk,
    const float* __restrict__ bin_, __bf16* __restrict__ ubf) {
  int s = blockIdx.z;
  int Ls = SEQLEN >> s;
  int mtiles = (BATCH * Ls) >> 4;
  int mt = blockIdx.x;
  if (mt >= mtiles) return;
  int wave = threadIdx.x >> 5, lane = threadIdx.x & 31;
  int nt = blockIdx.y * 4 + wave;             // 16 n-tiles
  int gm = mt * 16 + (lane & 15);
  int b = gm / Ls;
  int t = gm - b * Ls;
  long hrow = (long)b * SEQLEN + (long)t * (1 << s);
  const __bf16* wp = winupk + (long)s * (128 * 256);
  v8f acc = {};
  for (int kt = 0; kt < 4; ++kt) {
    v16bf a = load_a(hbf, hrow, DMODEL, kt * 32, lane);
    v16bf bb = load_b(wp, nt * 4 + kt, lane);
    acc = wmma_bf16(a, bb, acc);
  }
  int col = lane & 15, n0 = nt * 16, rb = (lane >> 4) * 8;
  float bv = bin_[s * 512 + n0 + col];
  long orow = (long)scale_rowoff(s) + mt * 16 + rb;
  for (int r = 0; r < 8; ++r)
    ubf[(orow + r) * DINNER + n0 + col] = (__bf16)(acc[r] + bv);
}

// ---------------------------------------------------------------------------
// Phase 3: dt = softplus(u@Wdt+bdt) * sigmoid(u@Wtau+btau), fused.
// Block = 8 waves; each wave owns one m-tile with its A-tiles held in
// registers (loaded once). Wdt/Wtau tiles staged through LDS by the Tensor
// Data Mover in 8 groups of 2 n-tiles (16KB + 16KB per group), consumed by
// all 8 waves via ds_load -> v_wmma.
// ---------------------------------------------------------------------------
__global__ __launch_bounds__(256) void dttau_kernel(
    const __bf16* __restrict__ ubf, const __bf16* __restrict__ wdtpk,
    const __bf16* __restrict__ wtaupk, const float* __restrict__ bdt,
    const float* __restrict__ btau, float* __restrict__ dtf) {
  int s = blockIdx.z;
  int Ls = SEQLEN >> s;
  int mtiles = (BATCH * Ls) >> 4;
  if (blockIdx.x * 8 >= (unsigned)mtiles) return;   // whole block uniform
  int wave = threadIdx.x >> 5, lane = threadIdx.x & 31;
  int mt = blockIdx.x * 8 + wave;
  int so = scale_rowoff(s);
  const __bf16* Ab = ubf + (long)so * DINNER;
  long row = (long)mt * 16 + (lane & 15);
  const __bf16* wd = wdtpk + (long)s * 65536;
  const __bf16* wt = wtaupk + (long)s * 65536;

  __shared__ __bf16 ldsW[16384];   // 32KB: [0..8191]=Wdt grp, [8192..]=Wtau grp
  unsigned lds0 = (unsigned)(size_t)(&ldsW[0]);

  // Preload this wave's 16x256 A panel into registers (8 k-tiles, 64 VGPRs).
  v16bf a[8];
  for (int kt = 0; kt < 8; ++kt) a[kt] = load_a(Ab, row, DINNER, kt * 32, lane);

  int col = lane & 15, rb = (lane >> 4) * 8;
  for (int g = 0; g < 8; ++g) {
    if (wave == 0) {
      tdm_load_b16(lds0,         wd + g * 8192, 8192);   // 2 n-tiles of Wdt
      tdm_load_b16(lds0 + 16384, wt + g * 8192, 8192);   // 2 n-tiles of Wtau
      __builtin_amdgcn_s_wait_tensorcnt(0);
    }
    __syncthreads();
    for (int j = 0; j < 2; ++j) {
      int nt = g * 2 + j;
      v8f accd = {}, acct = {};
      for (int kt = 0; kt < 8; ++kt) {
        v16bf bd_ = load_b_lds(&ldsW[(j * 8 + kt) * 512], lane);
        v16bf bt_ = load_b_lds(&ldsW[8192 + (j * 8 + kt) * 512], lane);
        accd = wmma_bf16(a[kt], bd_, accd);
        acct = wmma_bf16(a[kt], bt_, acct);
      }
      int n0 = nt * 16;
      float bd = bdt[s * 256 + n0 + col];
      float bt = btau[s * 256 + n0 + col];
      for (int r = 0; r < 8; ++r) {
        float xd = accd[r] + bd;
        float xt = acct[r] + bt;
        float sp = (xd > 20.f) ? xd : log1pf(__expf(xd));
        float sg = 1.f / (1.f + __expf(-xt));
        dtf[((long)so + mt * 16 + rb + r) * DINNER + n0 + col] = sp * sg;
      }
    }
    __syncthreads();
  }
}

// ---------------------------------------------------------------------------
// Phase 4: BC = u @ Wx + bx   (K=256, N=32) -> Bv cols 0..15, Cv cols 16..31
// ---------------------------------------------------------------------------
__global__ __launch_bounds__(64) void bc_kernel(
    const __bf16* __restrict__ ubf, const __bf16* __restrict__ wxpk,
    const float* __restrict__ bx, float* __restrict__ bcb) {
  int s = blockIdx.z;
  int Ls = SEQLEN >> s;
  int mtiles = (BATCH * Ls) >> 4;
  int mt = blockIdx.x;
  if (mt >= mtiles) return;
  int wave = threadIdx.x >> 5, lane = threadIdx.x & 31;
  int nt = wave;                              // 2 n-tiles
  int so = scale_rowoff(s);
  const __bf16* Ab = ubf + (long)so * DINNER;
  long row = (long)mt * 16 + (lane & 15);
  const __bf16* wx = wxpk + (long)s * 8192;
  v8f acc = {};
  for (int kt = 0; kt < 8; ++kt)
    acc = wmma_bf16(load_a(Ab, row, DINNER, kt * 32, lane),
                    load_b(wx, nt * 8 + kt, lane), acc);
  int col = lane & 15, n0 = nt * 16, rb = (lane >> 4) * 8;
  float bv = bx[s * 32 + n0 + col];
  for (int r = 0; r < 8; ++r)
    bcb[((long)so + mt * 16 + rb + r) * 32 + n0 + col] = acc[r] + bv;
}

// ---------------------------------------------------------------------------
// Phase 5: sequential liquid scan. One block per (scale,batch); one thread per
// (channel d, state n): 1 exp + 2 FMA per serial step. Finalize does the
// y = h.Cv reduction, z/silu, and the 256x128 out-projection at t = Ls-1.
// ---------------------------------------------------------------------------
__global__ __launch_bounds__(512) void scan_kernel(
    const float* __restrict__ dtf, const __bf16* __restrict__ ubf,
    const float* __restrict__ bcb, const __bf16* __restrict__ hbf,
    const float* __restrict__ A_log, const float* __restrict__ Dp,
    const float* __restrict__ Win, const float* __restrict__ bin_,
    const float* __restrict__ Wout, const float* __restrict__ bout,
    float* __restrict__ lastacc) {
  int s = blockIdx.x >> 3;
  int b = blockIdx.x & 7;
  int Ls = SEQLEN >> s;
  int stride = 1 << s;
  long base = (long)scale_rowoff(s) + (long)b * Ls;
  long lastrow = base + Ls - 1;
  int tid = threadIdx.x;
  int d = tid >> 4;
  int n = tid & 15;

  float a = -__expf(A_log[(s * DINNER + d) * NSTATE + n]);
  float h = 0.f, u_last = 0.f;
  for (int t = 0; t < Ls; ++t) {
    long row = base + t;
    long prow = row + 24;
    if (prow > lastrow) prow = lastrow;
    __builtin_prefetch(&dtf[prow * DINNER + d], 0, 1);   // global_prefetch_b8
    __builtin_prefetch(&ubf[prow * DINNER + d], 0, 1);
    float dtv = dtf[row * DINNER + d];
    float uv = (float)ubf[row * DINNER + d];
    float bv = bcb[row * 32 + n];
    h = __expf(dtv * a) * h + dtv * bv * uv;
    u_last = uv;
  }

  __shared__ float red[512];
  __shared__ float ybuf[DINNER];
  float cv = bcb[lastrow * 32 + 16 + n];
  red[tid] = h * cv;
  __syncthreads();
  if (n == 0) {
    float y = 0.f;
    for (int j = 0; j < NSTATE; ++j) y += red[d * 16 + j];
    y += u_last * Dp[s * DINNER + d];
    // z for the gate, only needed at the last timestep
    long hrow = (long)b * SEQLEN + (long)(Ls - 1) * stride;
    float z = bin_[s * 512 + 256 + d];
    for (int k = 0; k < DMODEL; ++k)
      z += (float)hbf[hrow * DMODEL + k] * Win[((long)s * DMODEL + k) * 512 + 256 + d];
    float sil = z / (1.f + __expf(-z));
    ybuf[d] = y * sil;
  }
  __syncthreads();
  if (tid < DMODEL) {
    float o = bout[s * DMODEL + tid];
    for (int dd = 0; dd < DINNER; ++dd)
      o += ybuf[dd] * Wout[((long)s * DINNER + dd) * DMODEL + tid];
    lastacc[(s * 8 + b) * DMODEL + tid] = o;
  }
}

// ---------------------------------------------------------------------------
// Phase 6: head — sum scales /3, layernorm, 128x64 relu MLP, 64x1 out
// ---------------------------------------------------------------------------
__global__ __launch_bounds__(128) void head_kernel(
    const float* __restrict__ lastacc, const float* __restrict__ ln_g,
    const float* __restrict__ ln_b, const float* __restrict__ W1,
    const float* __restrict__ b1, const float* __restrict__ W2,
    const float* __restrict__ b2, float* __restrict__ out) {
  int b = blockIdx.x, tid = threadIdx.x;
  __shared__ float sh[128];
  __shared__ float xln[128];
  __shared__ float h1[64];
  float v = (lastacc[b * 128 + tid] + lastacc[(8 + b) * 128 + tid] +
             lastacc[(16 + b) * 128 + tid]) * (1.f / 3.f);
  sh[tid] = v;
  __syncthreads();
  for (int off = 64; off > 0; off >>= 1) {
    if (tid < off) sh[tid] += sh[tid + off];
    __syncthreads();
  }
  float mu = sh[0] / 128.f;
  __syncthreads();
  float dv = v - mu;
  sh[tid] = dv * dv;
  __syncthreads();
  for (int off = 64; off > 0; off >>= 1) {
    if (tid < off) sh[tid] += sh[tid + off];
    __syncthreads();
  }
  float var = sh[0] / 128.f;
  __syncthreads();
  xln[tid] = dv * rsqrtf(var + 1e-5f) * ln_g[tid] + ln_b[tid];
  __syncthreads();
  if (tid < 64) {
    float acc = b1[tid];
    for (int k = 0; k < 128; ++k) acc += xln[k] * W1[k * 64 + tid];
    h1[tid] = fmaxf(acc, 0.f);
  }
  __syncthreads();
  if (tid == 0) {
    float o = b2[0];
    for (int j = 0; j < 64; ++j) o += h1[j] * W2[j];
    out[b] = o;
  }
}

// ---------------------------------------------------------------------------
// Host launcher
// ---------------------------------------------------------------------------
extern "C" void kernel_launch(void* const* d_in, const int* in_sizes, int n_in,
                              void* d_out, int out_size, void* d_ws,
                              size_t ws_size, hipStream_t stream) {
  const float* x     = (const float*)d_in[0];
  const float* Wp    = (const float*)d_in[1];
  const float* bp    = (const float*)d_in[2];
  const float* Win   = (const float*)d_in[3];
  const float* bin_  = (const float*)d_in[4];
  const float* Wdt   = (const float*)d_in[5];
  const float* bdt   = (const float*)d_in[6];
  const float* Wtau  = (const float*)d_in[7];
  const float* btau  = (const float*)d_in[8];
  const float* Wx    = (const float*)d_in[9];
  const float* bx    = (const float*)d_in[10];
  const float* A_log = (const float*)d_in[11];
  const float* Dp    = (const float*)d_in[12];
  const float* Wout  = (const float*)d_in[13];
  const float* bout  = (const float*)d_in[14];
  const float* ln_g  = (const float*)d_in[15];
  const float* ln_b  = (const float*)d_in[16];
  const float* W1    = (const float*)d_in[17];
  const float* b1    = (const float*)d_in[18];
  const float* W2    = (const float*)d_in[19];
  const float* b2    = (const float*)d_in[20];
  float* out = (float*)d_out;

  // Workspace carve-out (bytes, 256B aligned slices)
  char* ws = (char*)d_ws;
  size_t off = 0;
  auto carve = [&](size_t bytes) {
    size_t r = off;
    off = (off + bytes + 255) & ~(size_t)255;
    return r;
  };
  const size_t nBL = (size_t)BATCH * SEQLEN;
  __bf16* xbf    = (__bf16*)(ws + carve(nBL * FDIM * 2));
  __bf16* hbf    = (__bf16*)(ws + carve(nBL * DMODEL * 2));
  __bf16* wpk    = (__bf16*)(ws + carve((size_t)FDIM * DMODEL * 2));
  __bf16* winupk = (__bf16*)(ws + carve((size_t)NSCALE * 128 * 256 * 2));
  __bf16* wdtpk  = (__bf16*)(ws + carve((size_t)NSCALE * 256 * 256 * 2));
  __bf16* wtaupk = (__bf16*)(ws + carve((size_t)NSCALE * 256 * 256 * 2));
  __bf16* wxpk   = (__bf16*)(ws + carve((size_t)NSCALE * 256 * 32 * 2));
  __bf16* ubf    = (__bf16*)(ws + carve((size_t)TOTROWS * DINNER * 2));
  float*  dtf    = (float*)(ws + carve((size_t)TOTROWS * DINNER * 4));
  float*  bcb    = (float*)(ws + carve((size_t)TOTROWS * 32 * 4));
  float*  lacc   = (float*)(ws + carve((size_t)NSCALE * BATCH * DMODEL * 4));
  (void)ws_size; (void)in_sizes; (void)n_in; (void)out_size;

  // Convert activations + pre-swizzle weights to WMMA tile layout
  cvt_bf16_kernel<<<4096, 256, 0, stream>>>(x, xbf, (int)(nBL * FDIM));
  pack_w_kernel<<<32, 256, 0, stream>>>(Wp, wpk, FDIM, DMODEL, DMODEL);
  for (int s = 0; s < NSCALE; ++s) {
    pack_w_kernel<<<128, 256, 0, stream>>>(Win + (size_t)s * 128 * 512,
                                           winupk + (size_t)s * 128 * 256,
                                           128, 256, 512);
    pack_w_kernel<<<256, 256, 0, stream>>>(Wdt + (size_t)s * 65536,
                                           wdtpk + (size_t)s * 65536,
                                           256, 256, 256);
    pack_w_kernel<<<256, 256, 0, stream>>>(Wtau + (size_t)s * 65536,
                                           wtaupk + (size_t)s * 65536,
                                           256, 256, 256);
    pack_w_kernel<<<32, 256, 0, stream>>>(Wx + (size_t)s * 8192,
                                          wxpk + (size_t)s * 8192,
                                          256, 32, 32);
  }

  // WMMA GEMM phases
  proj_kernel<<<dim3(2048, 2), 128, 0, stream>>>(xbf, wpk, bp, hbf);
  uproj_kernel<<<dim3(2048, 4, NSCALE), 128, 0, stream>>>(hbf, winupk, bin_, ubf);
  dttau_kernel<<<dim3(256, 1, NSCALE), 256, 0, stream>>>(ubf, wdtpk, wtaupk,
                                                         bdt, btau, dtf);
  bc_kernel<<<dim3(2048, 1, NSCALE), 64, 0, stream>>>(ubf, wxpk, bx, bcb);

  // Sequential scan (24 independent recurrences) + finalize
  scan_kernel<<<NSCALE * BATCH, 512, 0, stream>>>(dtf, ubf, bcb, hbf, A_log, Dp,
                                                  Win, bin_, Wout, bout, lacc);
  // LN + MLP head
  head_kernel<<<BATCH, 128, 0, stream>>>(lacc, ln_g, ln_b, W1, b1, W2, b2, out);
}
